// MoELayer_41016937677061
// MI455X (gfx1250) — matrix operand order repaired and checked by reading, not compiled, not run on hardware
//
#include <hip/hip_runtime.h>
#include <hip/hip_bf16.h>
#include <math.h>

// Problem constants (from reference): B=1, S=2048, D=1024, F=4096, E=8, K=2.
#define MTOK  2048
#define DDIM  1024
#define FDIM  4096
#define NEXP  8
#define MK    (MTOK * 2)            // 4096 (token, expert) assignments
#define TM    128                   // rows per block tile (reuse lever for weights)
#define BN    64                    // cols per block tile
#define BK    32                    // k-depth per stage (= WMMA K)
#define BKP   40                    // padded LDS row (bank-conflict pad, keeps 16B align)
#define MKPAD (MK + NEXP * TM)      // 5120: per-expert segments padded to TM

typedef __attribute__((ext_vector_type(16))) __bf16 v16bf;
typedef __attribute__((ext_vector_type(8)))  __bf16 bf16x8;
typedef __attribute__((ext_vector_type(8)))  float  v8f;
typedef int vsi4 __attribute__((vector_size(16)));   // matches builtin's b128 payload type

#ifndef __has_builtin
#define __has_builtin(x) 0
#endif
#if __has_builtin(__builtin_amdgcn_global_load_async_to_lds_b128)
#define HAVE_ASYNC_LDS 1
#else
#define HAVE_ASYNC_LDS 0
#endif

// 16B global->LDS copy: async DMA (ASYNCcnt) when available, else through VGPRs.
__device__ __forceinline__ void copy16_g2l(const __bf16* gsrc, __bf16* ldst) {
#if HAVE_ASYNC_LDS
  __builtin_amdgcn_global_load_async_to_lds_b128(
      (__attribute__((address_space(1))) vsi4*)gsrc,
      (__attribute__((address_space(3))) vsi4*)ldst, 0, 0);
#else
  *(bf16x8*)ldst = *(const bf16x8*)gsrc;
#endif
}

__device__ __forceinline__ void wait_async_lds() {
#if HAVE_ASYNC_LDS
#if __has_builtin(__builtin_amdgcn_s_wait_asynccnt)
  __builtin_amdgcn_s_wait_asynccnt(0);
#else
  asm volatile("s_wait_asynccnt 0x0" ::: "memory");
#endif
#endif
}

__device__ __forceinline__ __bf16 f2bf(float f) { return (__bf16)f; }

__device__ __forceinline__ v16bf ldfrag(const __bf16* p0, const __bf16* p1) {
  bf16x8 lo = *(const bf16x8*)p0;
  bf16x8 hi = *(const bf16x8*)p1;
  v16bf f;
#pragma unroll
  for (int i = 0; i < 8; ++i) { f[i] = lo[i]; f[8 + i] = hi[i]; }
  return f;
}

__device__ __forceinline__ float gelu_tanh(float x) {
  float x3 = x * x * x;
  return 0.5f * x * (1.0f + tanhf(0.7978845608028654f * (x + 0.044715f * x3)));
}

// ---------------- Routing ----------------

__global__ void moe_init(int* counts, int* cursor) {
  int i = threadIdx.x;
  if (i < NEXP) { counts[i] = 0; cursor[i] = 0; }
}

__global__ __launch_bounds__(64) void moe_router(const float* __restrict__ x,
                                                 const float* __restrict__ rw,
                                                 float* __restrict__ gate,
                                                 int* __restrict__ eidx,
                                                 int* __restrict__ counts) {
  const int t = blockIdx.x;
  const int tid = threadIdx.x;
  float acc[NEXP];
#pragma unroll
  for (int e = 0; e < NEXP; ++e) acc[e] = 0.f;
  for (int i = tid; i < DDIM; i += 64) {
    float xv = x[(size_t)t * DDIM + i];
    const float* wr = rw + (size_t)i * NEXP;   // router_w is [D, E] row-major
#pragma unroll
    for (int e = 0; e < NEXP; ++e) acc[e] += xv * wr[e];
  }
  __shared__ float red[NEXP][64];
#pragma unroll
  for (int e = 0; e < NEXP; ++e) red[e][tid] = acc[e];
  __syncthreads();
  if (tid == 0) {
    float logit[NEXP];
    for (int e = 0; e < NEXP; ++e) {
      float s = 0.f;
      for (int j = 0; j < 64; ++j) s += red[e][j];
      logit[e] = s;
    }
    float mx = logit[0];
    for (int e = 1; e < NEXP; ++e) mx = fmaxf(mx, logit[e]);
    float p[NEXP], den = 0.f;
    for (int e = 0; e < NEXP; ++e) { p[e] = __expf(logit[e] - mx); den += p[e]; }
    for (int e = 0; e < NEXP; ++e) p[e] /= den;
    int i0 = 0;
    for (int e = 1; e < NEXP; ++e) if (p[e] > p[i0]) i0 = e;
    int i1 = (i0 == 0) ? 1 : 0;
    for (int e = 0; e < NEXP; ++e) if (e != i0 && p[e] > p[i1]) i1 = e;
    gate[2 * t] = p[i0];     eidx[2 * t] = i0;
    gate[2 * t + 1] = p[i1]; eidx[2 * t + 1] = i1;
    atomicAdd(&counts[i0], 1);
    atomicAdd(&counts[i1], 1);
  }
}

__global__ void moe_scan(const int* __restrict__ counts, int* __restrict__ off) {
  if (threadIdx.x == 0 && blockIdx.x == 0) {
    int acc = 0;
    off[0] = 0;
    for (int e = 0; e < NEXP; ++e) {
      acc += ((counts[e] + TM - 1) / TM) * TM;   // pad each segment to TM rows
      off[e + 1] = acc;
    }
  }
}

__global__ void moe_fill(const int* __restrict__ off, int* __restrict__ perm,
                         int* __restrict__ row_expert) {
  int p = blockIdx.x * blockDim.x + threadIdx.x;
  if (p >= MKPAD) return;
  perm[p] = -1;
  int e = NEXP - 1;
  for (int k = 0; k < NEXP; ++k) {
    if (p < off[k + 1]) { e = k; break; }
  }
  row_expert[p] = e;
}

__global__ void moe_scatter(const int* __restrict__ eidx, const int* __restrict__ off,
                            int* __restrict__ cursor, int* __restrict__ perm,
                            int* __restrict__ inv) {
  int r = blockIdx.x * blockDim.x + threadIdx.x;
  if (r >= MK) return;
  int e = eidx[r];
  int pos = off[e] + atomicAdd(&cursor[e], 1);
  perm[pos] = r;
  inv[r] = pos;
}

__global__ __launch_bounds__(256) void moe_gather(const float* __restrict__ x,
                                                  const int* __restrict__ perm,
                                                  __bf16* __restrict__ xg) {
  int p = blockIdx.x;
  int r = perm[p];
  if (r < 0) {
    for (int c = threadIdx.x; c < DDIM; c += 256) xg[(size_t)p * DDIM + c] = f2bf(0.f);
  } else {
    int t = r >> 1;
    for (int c = threadIdx.x; c < DDIM; c += 256)
      xg[(size_t)p * DDIM + c] = f2bf(x[(size_t)t * DDIM + c]);
  }
}

// ---------------- GEMM1: v = xg@(wv*sv), y = gelu(xg@(w*s)), H = v*y (bf16) ----------------
// Block: 256 threads = 8 waves in 4x2; tile TM(128) x BN(64); K staged in BK(32).

__global__ __launch_bounds__(256) void moe_gemm1(
    const __bf16* __restrict__ xg,
    const float* __restrict__ wv, const float* __restrict__ sv,
    const float* __restrict__ wg, const float* __restrict__ sg,
    const int* __restrict__ row_expert,
    __bf16* __restrict__ H) {
  const int rowBase = blockIdx.x * TM;
  const int colBase = blockIdx.y * BN;
  const int e = row_expert[rowBase];
  const size_t wbase = (size_t)e * DDIM * FDIM;

  __shared__ __align__(16) __bf16 lA[TM][BKP];
  __shared__ __align__(16) __bf16 lBv[BN][BKP];   // transposed: [n][k]
  __shared__ __align__(16) __bf16 lBw[BN][BKP];

  const int tid = threadIdx.x;
  const int lane = tid & 31;
  const int wid = tid >> 5;
  const int wm = wid >> 1, wn = wid & 1;   // 4x2 waves, each 32x32
  const int half = lane >> 4;
  const int l15 = lane & 15;

  v8f accv[2][2] = {};
  v8f accy[2][2] = {};

  for (int k0 = 0; k0 < DDIM; k0 += BK) {
    // A tile (TMxBK bf16): async DMA global->LDS, overlapped with B conversion below.
#pragma unroll
    for (int it = 0; it < 2; ++it) {
      int c = tid + it * 256;                 // 512 chunks of 8 bf16
      int rr = c >> 2, jj = (c & 3) << 3;
      copy16_g2l(&xg[(size_t)(rowBase + rr) * DDIM + k0 + jj], &lA[rr][jj]);
    }
    // B tiles (BKxBN fp32 -> bf16, transposed [n][k]): one chunk per thread.
    {
      int kk = tid >> 3, nn = (tid & 7) << 3;
      size_t gi = wbase + (size_t)(k0 + kk) * FDIM + colBase + nn;
      float4 a0 = *(const float4*)&wv[gi];
      float4 a1 = *(const float4*)&wv[gi + 4];
      float4 b0 = *(const float4*)&wg[gi];
      float4 b1 = *(const float4*)&wg[gi + 4];
      lBv[nn + 0][kk] = f2bf(a0.x); lBv[nn + 1][kk] = f2bf(a0.y);
      lBv[nn + 2][kk] = f2bf(a0.z); lBv[nn + 3][kk] = f2bf(a0.w);
      lBv[nn + 4][kk] = f2bf(a1.x); lBv[nn + 5][kk] = f2bf(a1.y);
      lBv[nn + 6][kk] = f2bf(a1.z); lBv[nn + 7][kk] = f2bf(a1.w);
      lBw[nn + 0][kk] = f2bf(b0.x); lBw[nn + 1][kk] = f2bf(b0.y);
      lBw[nn + 2][kk] = f2bf(b0.z); lBw[nn + 3][kk] = f2bf(b0.w);
      lBw[nn + 4][kk] = f2bf(b1.x); lBw[nn + 5][kk] = f2bf(b1.y);
      lBw[nn + 6][kk] = f2bf(b1.z); lBw[nn + 7][kk] = f2bf(b1.w);
    }
    if (k0 + BK < DDIM)
      __builtin_prefetch(&wv[wbase + (size_t)(k0 + BK) * FDIM + colBase], 0, 1);
    wait_async_lds();
    __syncthreads();

    v16bf af[2], bvf[2], bwf[2];
#pragma unroll
    for (int sm = 0; sm < 2; ++sm) {
      int rr = wm * 32 + sm * 16 + l15;
      // A layout: lanes<16 hold K {0..7,16..23}; lanes>=16 hold K {8..15,24..31}
      af[sm] = ldfrag(&lA[rr][half * 8], &lA[rr][16 + half * 8]);
    }
#pragma unroll
    for (int sn = 0; sn < 2; ++sn) {
      int cc = wn * 32 + sn * 16 + l15;
      // B layout: lanes<16 hold K {0..7,16..23}; lanes>=16 hold K {8..15,24..31}
      bvf[sn] = ldfrag(&lBv[cc][half * 8], &lBv[cc][16 + half * 8]);
      bwf[sn] = ldfrag(&lBw[cc][half * 8], &lBw[cc][16 + half * 8]);
    }
#pragma unroll
    for (int sm = 0; sm < 2; ++sm)
#pragma unroll
      for (int sn = 0; sn < 2; ++sn) {
        accv[sm][sn] = __builtin_amdgcn_wmma_f32_16x16x32_bf16(
            false, af[sm], false, bvf[sn], (short)0, accv[sm][sn], false, false);
        accy[sm][sn] = __builtin_amdgcn_wmma_f32_16x16x32_bf16(
            false, af[sm], false, bwf[sn], (short)0, accy[sm][sn], false, false);
      }
    __syncthreads();
  }

  // Epilogue: per-column dequant scales, v * gelu(y) -> bf16 H
#pragma unroll
  for (int sm = 0; sm < 2; ++sm)
#pragma unroll
    for (int sn = 0; sn < 2; ++sn) {
      int gn = colBase + wn * 32 + sn * 16 + l15;
      int gm = rowBase + wm * 32 + sm * 16 + half * 8;  // C: VGPR r -> M = r + half*8
      float scv = sv[(size_t)e * FDIM + gn];
      float scy = sg[(size_t)e * FDIM + gn];
#pragma unroll
      for (int r = 0; r < 8; ++r) {
        float v = accv[sm][sn][r] * scv;
        float y = gelu_tanh(accy[sm][sn][r] * scy);
        H[(size_t)(gm + r) * FDIM + gn] = f2bf(v * y);
      }
    }
}

// ---------------- GEMM2: op = H @ (w1*s1) (fp32 partials, permuted rows) ----------------

__global__ __launch_bounds__(256) void moe_gemm2(
    const __bf16* __restrict__ H,
    const float* __restrict__ w1, const float* __restrict__ s1,
    const int* __restrict__ row_expert,
    float* __restrict__ op) {
  const int rowBase = blockIdx.x * TM;
  const int colBase = blockIdx.y * BN;
  const int e = row_expert[rowBase];
  const size_t wbase = (size_t)e * FDIM * DDIM;

  __shared__ __align__(16) __bf16 lA[TM][BKP];
  __shared__ __align__(16) __bf16 lB[BN][BKP];

  const int tid = threadIdx.x;
  const int lane = tid & 31;
  const int wid = tid >> 5;
  const int wm = wid >> 1, wn = wid & 1;
  const int half = lane >> 4;
  const int l15 = lane & 15;

  v8f acc[2][2] = {};

  for (int k0 = 0; k0 < FDIM; k0 += BK) {
#pragma unroll
    for (int it = 0; it < 2; ++it) {
      int c = tid + it * 256;
      int rr = c >> 2, jj = (c & 3) << 3;
      copy16_g2l(&H[(size_t)(rowBase + rr) * FDIM + k0 + jj], &lA[rr][jj]);
    }
    {
      int kk = tid >> 3, nn = (tid & 7) << 3;
      size_t gi = wbase + (size_t)(k0 + kk) * DDIM + colBase + nn;
      float4 a0 = *(const float4*)&w1[gi];
      float4 a1 = *(const float4*)&w1[gi + 4];
      lB[nn + 0][kk] = f2bf(a0.x); lB[nn + 1][kk] = f2bf(a0.y);
      lB[nn + 2][kk] = f2bf(a0.z); lB[nn + 3][kk] = f2bf(a0.w);
      lB[nn + 4][kk] = f2bf(a1.x); lB[nn + 5][kk] = f2bf(a1.y);
      lB[nn + 6][kk] = f2bf(a1.z); lB[nn + 7][kk] = f2bf(a1.w);
    }
    if (k0 + BK < FDIM)
      __builtin_prefetch(&w1[wbase + (size_t)(k0 + BK) * DDIM + colBase], 0, 1);
    wait_async_lds();
    __syncthreads();

    v16bf af[2], bf[2];
#pragma unroll
    for (int sm = 0; sm < 2; ++sm) {
      int rr = wm * 32 + sm * 16 + l15;
      af[sm] = ldfrag(&lA[rr][half * 8], &lA[rr][16 + half * 8]);
    }
#pragma unroll
    for (int sn = 0; sn < 2; ++sn) {
      int cc = wn * 32 + sn * 16 + l15;
      bf[sn] = ldfrag(&lB[cc][half * 8], &lB[cc][16 + half * 8]);
    }
#pragma unroll
    for (int sm = 0; sm < 2; ++sm)
#pragma unroll
      for (int sn = 0; sn < 2; ++sn)
        acc[sm][sn] = __builtin_amdgcn_wmma_f32_16x16x32_bf16(
            false, af[sm], false, bf[sn], (short)0, acc[sm][sn], false, false);
    __syncthreads();
  }

#pragma unroll
  for (int sm = 0; sm < 2; ++sm)
#pragma unroll
    for (int sn = 0; sn < 2; ++sn) {
      int gn = colBase + wn * 32 + sn * 16 + l15;
      int gm = rowBase + wm * 32 + sm * 16 + half * 8;
      float sc = s1[(size_t)e * DDIM + gn];
#pragma unroll
      for (int r = 0; r < 8; ++r)
        op[(size_t)(gm + r) * DDIM + gn] = acc[sm][sn][r] * sc;
    }
}

// ---------------- Gate combine (deterministic, no fp atomics) ----------------

__global__ __launch_bounds__(256) void moe_combine(const float* __restrict__ op,
                                                   const float* __restrict__ gate,
                                                   const int* __restrict__ inv,
                                                   float* __restrict__ out) {
  int t = blockIdx.x;
  int p0 = inv[2 * t], p1 = inv[2 * t + 1];
  float g0 = gate[2 * t], g1 = gate[2 * t + 1];
  for (int c = threadIdx.x; c < DDIM; c += 256)
    out[(size_t)t * DDIM + c] =
        g0 * op[(size_t)p0 * DDIM + c] + g1 * op[(size_t)p1 * DDIM + c];
}

// ---------------- Launch ----------------

extern "C" void kernel_launch(void* const* d_in, const int* in_sizes, int n_in,
                              void* d_out, int out_size, void* d_ws, size_t ws_size,
                              hipStream_t stream) {
  const float* x  = (const float*)d_in[0];   // [1,2048,1024]
  const float* rw = (const float*)d_in[1];   // [1024,8]
  const float* wv = (const float*)d_in[2];   // [8,1024,4096]
  const float* sv = (const float*)d_in[3];   // [8,1,4096]
  const float* wg = (const float*)d_in[4];   // [8,1024,4096]
  const float* sg = (const float*)d_in[5];   // [8,1,4096]
  const float* w1 = (const float*)d_in[6];   // [8,4096,1024]
  const float* s1 = (const float*)d_in[7];   // [8,1,1024]
  float* out = (float*)d_out;
  (void)in_sizes; (void)n_in; (void)out_size; (void)ws_size;

  char* ws = (char*)d_ws;
  size_t o = 0;
  auto take = [&](size_t bytes) -> char* {
    char* p = ws + o;
    o += (bytes + 255) & ~(size_t)255;
    return p;
  };
  int*    counts = (int*)take(NEXP * 4);
  int*    cursor = (int*)take(NEXP * 4);
  int*    off    = (int*)take((NEXP + 1) * 4);
  float*  gate   = (float*)take((size_t)MK * 4);
  int*    eidx   = (int*)take((size_t)MK * 4);
  int*    perm   = (int*)take((size_t)MKPAD * 4);
  int*    inv    = (int*)take((size_t)MK * 4);
  int*    rowe   = (int*)take((size_t)MKPAD * 4);
  __bf16* xg     = (__bf16*)take((size_t)MKPAD * DDIM * 2);
  __bf16* H      = (__bf16*)take((size_t)MKPAD * FDIM * 2);
  float*  op     = (float*)take((size_t)MKPAD * DDIM * 4);

  moe_init<<<1, 64, 0, stream>>>(counts, cursor);
  moe_router<<<MTOK, 64, 0, stream>>>(x, rw, gate, eidx, counts);
  moe_scan<<<1, 1, 0, stream>>>(counts, off);
  moe_fill<<<(MKPAD + 255) / 256, 256, 0, stream>>>(off, perm, rowe);
  moe_scatter<<<MK / 256, 256, 0, stream>>>(eidx, off, cursor, perm, inv);
  moe_gather<<<MKPAD, 256, 0, stream>>>(x, perm, xg);
  dim3 g1(MKPAD / TM, FDIM / BN);
  moe_gemm1<<<g1, 256, 0, stream>>>(xg, wv, sv, wg, sg, rowe, H);
  dim3 g2(MKPAD / TM, DDIM / BN);
  moe_gemm2<<<g2, 256, 0, stream>>>(H, w1, s1, rowe, op);
  moe_combine<<<MTOK, 256, 0, stream>>>(op, gate, inv, out);
}
